// WindowAttention3D_5282809774674
// MI455X (gfx1250) — matrix operand order repaired and checked by reading, not compiled, not run on hardware
//
#include <hip/hip_runtime.h>

typedef __attribute__((ext_vector_type(16))) _Float16 v16h;
typedef __attribute__((ext_vector_type(8)))  _Float16 half8;
typedef __attribute__((ext_vector_type(8)))  float    v8f;

#define DIM   256
#define HEADS 8
#define HD    32
#define N_TOK 98
#define NP    112   // rows padded to multiple of 16
#define KP    128   // key-dim padding for P / Vt (multiple of 32)
#define NWMSK 64

struct Smem {
  union R0 {
    _Float16 X[NP * DIM];                                    // 57,344 B (QKV stage)
    struct SP { float S[NP * NP]; _Float16 P[NP * KP]; } sp; // 50,176 + 28,672 B
  } r0;                                                      // 78,848 B
  _Float16 Q[NP * DIM];    // 57,344 B (pre-scaled)
  _Float16 K[NP * DIM];    // 57,344 B
  _Float16 Vt[DIM * KP];   // 65,536 B (transposed: [channel][row])
  _Float16 A[NP * DIM];    // 57,344 B (attention output, f16)
};                         // total 316,416 B (<= 320KB WGP LDS)

union AFrag { v16h v; _Float16 h[16]; half8 q[2]; };

__device__ inline v8f wmma_f16(v16h a, v16h b, v8f c) {
  return __builtin_amdgcn_wmma_f32_16x16x32_f16(
      /*neg_a=*/false, a, /*neg_b=*/false, b,
      /*c_mod=*/(short)0, c, /*reuse_a=*/false, /*reuse_b=*/false);
}

// A-fragment (16x32 f16): lanes<16 hold K=0..7,16..23; lanes>=16 hold K=8..15,24..31.
// caller passes base already offset by (lane>=16 ? 8 : 0); chunks at +0 and +16.
__device__ inline v16h load_a_frag(const _Float16* base) {
  AFrag f;
  f.q[0] = *(const half8*)(base);
  f.q[1] = *(const half8*)(base + 16);
  return f.v;
}

// B-fragment (32x16 f16): lane n holds column n, K=0..7,16..23 (lanes<16) or 8..15,24..31.
// caller passes base already offset by (lane>=16 ? 8 : 0); chunks at +0 and +16.
__device__ inline v16h load_b_frag(const _Float16* base) {
  AFrag f;
  f.q[0] = *(const half8*)(base);
  f.q[1] = *(const half8*)(base + 16);
  return f.v;
}

// ---- Prep: transpose + convert fp32 weights to f16 column-major in d_ws ----
__global__ __launch_bounds__(256)
void prep_weights_kernel(const float* __restrict__ qkv_w,
                         const float* __restrict__ proj_w,
                         _Float16* __restrict__ wq, _Float16* __restrict__ wp) {
  const int total = 3 * DIM * DIM + DIM * DIM;
  for (int idx = blockIdx.x * blockDim.x + threadIdx.x; idx < total;
       idx += gridDim.x * blockDim.x) {
    if (idx < 3 * DIM * DIM) {
      int col = idx / DIM, k = idx % DIM;
      wq[idx] = (_Float16)qkv_w[k * (3 * DIM) + col];
    } else {
      int j = idx - 3 * DIM * DIM;
      int col = j / DIM, k = j % DIM;
      wp[j] = (_Float16)proj_w[k * DIM + col];
    }
  }
}

__global__ __launch_bounds__(256)
void win_attn3d_kernel(const float* __restrict__ x, const float* __restrict__ mask,
                       const _Float16* __restrict__ wq,   // qkv weights f16 [768][256]
                       const float* __restrict__ qkv_b,
                       const _Float16* __restrict__ wp,   // proj weights f16 [256][256]
                       const float* __restrict__ proj_b,
                       const float* __restrict__ bias_table,
                       const int* __restrict__ rel_index,
                       float* __restrict__ out) {
  __shared__ Smem sm;
  const int tid    = threadIdx.x;
  const int lane   = tid & 31;
  const int wave   = tid >> 5;
  const int b      = blockIdx.x;
  const int wid    = b & (NWMSK - 1);       // window index within mask
  const int lrow   = lane & 15;             // N / M-sub index inside fragments
  const int hiHalf = lane >> 4;             // 0 or 1
  const float scale = 0.17677669529663687f; // 32^-0.5

  // ---- Stage 0: x -> LDS f16 (rows >= 98 zeroed); zero Vt pad rows 112..127
  for (int i = tid; i < NP * (DIM / 4); i += 256) {
    int row = i / (DIM / 4);
    int c4  = (i % (DIM / 4)) * 4;
    _Float16 h0 = 0, h1 = 0, h2 = 0, h3 = 0;
    if (row < N_TOK) {
      const float4 xv = *(const float4*)(x + ((size_t)b * N_TOK + row) * DIM + c4);
      h0 = (_Float16)xv.x; h1 = (_Float16)xv.y;
      h2 = (_Float16)xv.z; h3 = (_Float16)xv.w;
    }
    _Float16* d = &sm.r0.X[row * DIM + c4];
    d[0] = h0; d[1] = h1; d[2] = h2; d[3] = h3;
  }
  for (int i = tid; i < DIM * (KP - NP); i += 256) {
    int c = i / (KP - NP);
    int t = i % (KP - NP);
    sm.Vt[c * KP + NP + t] = (_Float16)0.f;
  }
  __syncthreads();

  // ---- Stage 1: QKV = X @ qkv_w + qkv_b (112 x 768), route into Q / K / Vt.
  // Wave owns a column tile; B-fragments for all 8 K-steps live in registers
  // and are reused across the 7 row tiles. Q/K/V routing is uniform in ct
  // (scalar branch, no EXEC masking); V writeback packs to one b128 store.
  for (int ct = wave; ct < 48; ct += 8) {
    const int cg = ct * 16 + lrow;
    v16h bf[8];
#pragma unroll
    for (int ks = 0; ks < 8; ++ks)
      bf[ks] = load_b_frag(&wq[(size_t)cg * DIM + ks * 32 + hiHalf * 8]);
    const float bias = qkv_b[cg];
    for (int rt = 0; rt < 7; ++rt) {
      const int arow = rt * 16 + lrow;
      v8f acc = {};
#pragma unroll
      for (int ks = 0; ks < 8; ++ks) {
        v16h a = load_a_frag(&sm.r0.X[arow * DIM + ks * 32 + hiHalf * 8]);
        acc = wmma_f16(a, bf[ks], acc);
      }
      if (ct < 16) {                      // Q tile (pre-scaled)
#pragma unroll
        for (int r = 0; r < 8; ++r) {
          int m = rt * 16 + r + hiHalf * 8;
          sm.Q[m * DIM + cg] = (_Float16)((acc[r] + bias) * scale);
        }
      } else if (ct < 32) {               // K tile
#pragma unroll
        for (int r = 0; r < 8; ++r) {
          int m = rt * 16 + r + hiHalf * 8;
          sm.K[m * DIM + (cg - DIM)] = (_Float16)(acc[r] + bias);
        }
      } else {                            // V tile -> transposed, contiguous pack
        half8 hv;
#pragma unroll
        for (int r = 0; r < 8; ++r) hv[r] = (_Float16)(acc[r] + bias);
        *(half8*)&sm.Vt[(cg - 2 * DIM) * KP + rt * 16 + hiHalf * 8] = hv;
      }
    }
  }
  __syncthreads();

  // zero P pad columns 112..127 once (X region now dead)
  for (int i = tid; i < NP * (KP - NP); i += 256) {
    int row = i / (KP - NP);
    int t   = i % (KP - NP);
    sm.r0.sp.P[row * KP + NP + t] = (_Float16)0.f;
  }
  __syncthreads();

  // ---- Stage 2: per-head attention
  const int rg = tid >> 4;   // softmax: row group 0..15
  const int sl = tid & 15;   // softmax: sub-lane 0..15
  for (int h = 0; h < HEADS; ++h) {
    // S = Q_h @ K_h^T + rel_bias + mask (one 16x16x32 WMMA per tile, K = hd)
    for (int job = wave; job < 49; job += 8) {
      int srt = job / 7, sct = job % 7;
      int qrow = srt * 16 + lrow;
      int krow = sct * 16 + lrow;
      v16h a  = load_a_frag(&sm.Q[qrow * DIM + h * HD + hiHalf * 8]);
      v16h bb = load_b_frag(&sm.K[krow * DIM + h * HD + hiHalf * 8]);
      v8f acc = {};
      acc = wmma_f16(a, bb, acc);
#pragma unroll
      for (int r = 0; r < 8; ++r) {
        int m = srt * 16 + r + hiHalf * 8;
        int n = sct * 16 + lrow;
        float sval = acc[r];
        if (n < N_TOK && m < N_TOK) {
          int ridx = rel_index[m * N_TOK + n];
          sval += bias_table[ridx * HEADS + h] +
                  mask[((size_t)wid * N_TOK + m) * N_TOK + n];
        } else if (n >= N_TOK) {
          sval = -1e30f;   // padded key columns -> softmax weight 0
        }
        sm.r0.sp.S[m * NP + n] = sval;
      }
    }
    __syncthreads();

    // row softmax (fp32) -> P (f16): 16 lanes per row, shuffle-tree reductions
    for (int p = 0; p < 7; ++p) {
      const int row = p * 16 + rg;                 // 0..111
      const float* srow = &sm.r0.sp.S[row * NP];
      if (row < N_TOK) {
        float vals[7];
        float mx = -1e30f;
#pragma unroll
        for (int j = 0; j < 7; ++j) {
          vals[j] = srow[sl + 16 * j];
          mx = fmaxf(mx, vals[j]);
        }
#pragma unroll
        for (int d = 1; d < 16; d <<= 1) mx = fmaxf(mx, __shfl_xor(mx, d, 32));
        float sum = 0.f;
#pragma unroll
        for (int j = 0; j < 7; ++j) {
          vals[j] = __expf(vals[j] - mx);          // padded cols: exp(-huge)=0
          sum += vals[j];
        }
#pragma unroll
        for (int d = 1; d < 16; d <<= 1) sum += __shfl_xor(sum, d, 32);
        const float inv = 1.f / sum;
#pragma unroll
        for (int j = 0; j < 7; ++j)
          sm.r0.sp.P[row * KP + sl + 16 * j] = (_Float16)(vals[j] * inv);
      } else {
#pragma unroll
        for (int j = 0; j < 7; ++j)
          sm.r0.sp.P[row * KP + sl + 16 * j] = (_Float16)0.f;
      }
    }
    __syncthreads();

    // O = P @ V_h  -> A[:, h*32 .. h*32+31] (Vt gives contiguous B-fragments)
    for (int job = wave; job < 14; job += 8) {
      int ort = job % 7, oct = job / 7;
      int prow = ort * 16 + lrow;
      int vcol = h * HD + oct * 16 + lrow;
      v8f acc = {};
#pragma unroll
      for (int ks = 0; ks < 4; ++ks) {
        int k0 = ks * 32;
        v16h a  = load_a_frag(&sm.r0.sp.P[prow * KP + k0 + hiHalf * 8]);
        v16h bb = load_b_frag(&sm.Vt[vcol * KP + k0 + hiHalf * 8]);
        acc = wmma_f16(a, bb, acc);
      }
#pragma unroll
      for (int r = 0; r < 8; ++r) {
        int m = ort * 16 + r + hiHalf * 8;
        sm.A[m * DIM + h * HD + oct * 16 + lrow] = (_Float16)acc[r];
      }
    }
    __syncthreads();   // protect S/P before next head overwrites them
  }

  // ---- Stage 3: out = A @ proj_w + proj_b (B-fragments hoisted per col tile)
  for (int ct = wave; ct < 16; ct += 8) {
    const int cg = ct * 16 + lrow;
    v16h bf[8];
#pragma unroll
    for (int ks = 0; ks < 8; ++ks)
      bf[ks] = load_b_frag(&wp[(size_t)cg * DIM + ks * 32 + hiHalf * 8]);
    const float pb = proj_b[cg];
    for (int rt = 0; rt < 7; ++rt) {
      const int arow = rt * 16 + lrow;
      v8f acc = {};
#pragma unroll
      for (int ks = 0; ks < 8; ++ks) {
        v16h a = load_a_frag(&sm.A[arow * DIM + ks * 32 + hiHalf * 8]);
        acc = wmma_f16(a, bf[ks], acc);
      }
#pragma unroll
      for (int r = 0; r < 8; ++r) {
        int m = rt * 16 + r + hiHalf * 8;
        if (m < N_TOK)
          out[((size_t)b * N_TOK + m) * DIM + cg] = acc[r] + pb;
      }
    }
  }
}

extern "C" void kernel_launch(void* const* d_in, const int* in_sizes, int n_in,
                              void* d_out, int out_size, void* d_ws, size_t ws_size,
                              hipStream_t stream) {
  (void)n_in; (void)ws_size; (void)out_size;
  const float* x          = (const float*)d_in[0];
  const float* mask       = (const float*)d_in[1];
  const float* qkv_w      = (const float*)d_in[2];
  const float* qkv_b      = (const float*)d_in[3];
  const float* proj_w     = (const float*)d_in[4];
  const float* proj_b     = (const float*)d_in[5];
  const float* bias_table = (const float*)d_in[6];
  const int*   rel_index  = (const int*)d_in[7];
  float* out = (float*)d_out;

  _Float16* wqh = (_Float16*)d_ws;                      // 3*256*256 f16
  _Float16* wph = wqh + 3 * DIM * DIM;                  // 256*256 f16 (512KB total)

  hipLaunchKernelGGL(prep_weights_kernel, dim3(256), dim3(256), 0, stream,
                     qkv_w, proj_w, wqh, wph);

  const int Bw = in_sizes[0] / (N_TOK * DIM);           // 2048 windows
  hipLaunchKernelGGL(win_attn3d_kernel, dim3(Bw), dim3(256), 0, stream,
                     x, mask, wqh, qkv_b, wph, proj_b, bias_table,
                     rel_index, out);
}
